// SpatialAttention_19988777795755
// MI455X (gfx1250) — matrix operand order repaired and checked by reading, not compiled, not run on hardware
//
#include <hip/hip_runtime.h>

typedef __attribute__((ext_vector_type(16))) __bf16 v16bf;
typedef __attribute__((ext_vector_type(8)))  float  v8f;
typedef __attribute__((ext_vector_type(4)))  unsigned int u32x4;
typedef __attribute__((ext_vector_type(8)))  int i32x8;
typedef __attribute__((ext_vector_type(4)))  int i32x4;

#define T_DIM 96
#define C_DIM 128
#define HD    32
#define NHEAD 4

// region A: [0, 49152)        x staged as f32 (TDM destination), later 96x96 f32 scores
// region B: [49152, 73728)    x as bf16, later attention output (bf16)
// region C: [73728, 147456)   Q, K, V bf16 (96x128 each)
#define OFF_B 49152
#define OFF_C 73728
#define SMEM_BYTES 147456

__device__ __forceinline__ v8f wmma_bf16(v16bf a, v16bf b, v8f c) {
  // emits v_wmma_f32_16x16x32_bf16
  return __builtin_amdgcn_wmma_f32_16x16x32_bf16(false, a, false, b, (short)0, c,
                                                 false, false);
}

// A fragment (16x32, bf16). Lane holds row m; element e -> K = kk + 16*hi + e
__device__ __forceinline__ v16bf ld_a_bf(const __bf16* row, int kk, int hi) {
  const __bf16* p = row + kk + (hi ? 16 : 0);
  v16bf a;
#pragma unroll
  for (int e = 0; e < 8; ++e) a[e] = p[e];
#pragma unroll
  for (int e = 0; e < 8; ++e) a[e + 8] = p[8 + e];
  return a;
}

// A fragment gathered from f32 LDS (scores) with hardware f32->bf16 convert
__device__ __forceinline__ v16bf ld_a_f32(const float* row, int kk, int hi) {
  const float* p = row + kk + (hi ? 16 : 0);
  v16bf a;
#pragma unroll
  for (int e = 0; e < 8; ++e) a[e] = (__bf16)p[e];
#pragma unroll
  for (int e = 0; e < 8; ++e) a[e + 8] = (__bf16)p[8 + e];
  return a;
}

// B fragment (32x16). Lane holds column n; element e -> K = k0 + 16*hi + e.
// p must already point at [k0 + hi*16] for this lane's column; contiguous in K.
__device__ __forceinline__ v16bf ld_b_contig_bf(const __bf16* p) {
  v16bf b;
#pragma unroll
  for (int e = 0; e < 16; ++e) b[e] = p[e];
  return b;
}

__device__ __forceinline__ v16bf ld_b_contig_f32(const float* p) {
  v16bf b;
#pragma unroll
  for (int e = 0; e < 16; ++e) b[e] = (__bf16)p[e];
  return b;
}

// B fragment with K-stride (V[k][d] walking k)
__device__ __forceinline__ v16bf ld_b_strided_bf(const __bf16* p, int stride) {
  v16bf b;
#pragma unroll
  for (int e = 0; e < 16; ++e) b[e] = p[e * stride];
  return b;
}

// One 96x128 = X(96x128) @ W^T(128x128) + bias projection, result bf16 into LDS.
// Separate call per weight keeps the LDS destination's address space inferable.
__device__ __forceinline__ void proj128_lds(const float* __restrict__ W,
                                            const float* __restrict__ bias,
                                            const __bf16* X, __bf16* Y,
                                            int wv, int ln, int hi) {
  for (int t = wv; t < 48; t += 8) {
    const int m0  = (t >> 3) * 16;
    const int n0  = (t & 7) * 16;
    const int col = n0 + ln;
    const float bval = bias[col];
    v8f c;
#pragma unroll
    for (int r = 0; r < 8; ++r) c[r] = bval;
    const __bf16* arow = X + (m0 + ln) * C_DIM;
    const float*  wrow = W + (long)col * C_DIM;   // B[k][n] = W[n][k]
#pragma unroll
    for (int kk = 0; kk < C_DIM; kk += 32) {
      v16bf a = ld_a_bf(arow, kk, hi);
      v16bf b = ld_b_contig_f32(wrow + kk + hi * 16);
      c = wmma_bf16(a, b, c);
    }
    const int mrow = m0 + hi * 8;
#pragma unroll
    for (int r = 0; r < 8; ++r) Y[(mrow + r) * C_DIM + col] = (__bf16)c[r];
  }
}

__global__ __launch_bounds__(256)
void spatial_attn_fused(const float* __restrict__ x,
                        const float* __restrict__ Wq, const float* __restrict__ bq,
                        const float* __restrict__ Wk, const float* __restrict__ bk,
                        const float* __restrict__ Wv, const float* __restrict__ bv,
                        const float* __restrict__ Wo, const float* __restrict__ bo,
                        float* __restrict__ out) {
  extern __shared__ char smem_raw[];
  float*  s_xf = (float*)smem_raw;                  // 96x128 f32 (TDM landing zone)
  float*  s_sc = (float*)smem_raw;                  // 96x96 f32 scores (reuses region A)
  __bf16* s_x  = (__bf16*)(smem_raw + OFF_B);       // 96x128 bf16 (x, then attn-out)
  __bf16* s_q  = (__bf16*)(smem_raw + OFF_C);       // 96x128 bf16
  __bf16* s_k  = s_q + T_DIM * C_DIM;
  __bf16* s_v  = s_k + T_DIM * C_DIM;

  const int tid  = threadIdx.x;
  const int wv   = tid >> 5;        // wave id 0..7
  const int lane = tid & 31;
  const int ln   = lane & 15;
  const int hi   = lane >> 4;       // 0: lanes 0-15, 1: lanes 16-31
  const long node = blockIdx.x;
  const float* xb = x   + node * (long)(T_DIM * C_DIM);
  float*       ob = out + node * (long)(T_DIM * C_DIM);

  // ---- Phase 0: stage x -> LDS ----
#if __has_builtin(__builtin_amdgcn_tensor_load_to_lds) && __has_builtin(__builtin_amdgcn_s_wait_tensorcnt)
  // Tensor Data Mover: one wave DMA's the contiguous 96x128 f32 tile into LDS.
  if (wv == 0) {
    unsigned long long ga = (unsigned long long)(const void*)xb;
    // D# group0: count=1 | lds_addr=0 | global_addr[56:0] | type=2
    u32x4 g0 = { 1u, 0u,
                 (unsigned)(ga & 0xFFFFFFFFu),
                 (unsigned)(((ga >> 32) & 0x01FFFFFFu) | 0x80000000u) };
    // D# group1: data_size=4B; tensor_dim0=128, tensor_dim1=96;
    //            tile_dim0=128, tile_dim1=96; tensor_dim0_stride=128; dim1_stride=12288
    i32x8 g1 = { 0x00020000,
                 (int)(128u << 16),     // tensor_dim0[15:0] in bits 63:48
                 (int)(96u  << 16),     // tensor_dim1[15:0] in bits 95:80
                 (int)(128u << 16),     // tile_dim0 in bits 127:112
                 96,                    // tile_dim1 in bits 143:128
                 128,                   // tensor_dim0_stride[31:0]
                 0x30000000,            // tensor_dim1_stride[15:0]=12288 in bits 223:208
                 0 };
    i32x4 gz4 = {0, 0, 0, 0};
    i32x8 gz8 = {0, 0, 0, 0, 0, 0, 0, 0};
    // clang-23 / therock-10.0 lane: 6-arg form (extra int32x8 group before cpol)
    __builtin_amdgcn_tensor_load_to_lds(g0, g1, gz4, gz4, gz8, 0);
  }
  __builtin_amdgcn_s_wait_tensorcnt(0);
  __syncthreads();
  for (int i = tid; i < T_DIM * C_DIM; i += 256) s_x[i] = (__bf16)s_xf[i];
#else
  for (int i = tid; i < T_DIM * C_DIM; i += 256) s_x[i] = (__bf16)xb[i];
#endif
  __syncthreads();

  // ---- Phase 1: Q/K/V = x @ W^T + b ----
  proj128_lds(Wq, bq, s_x, s_q, wv, ln, hi);
  proj128_lds(Wk, bk, s_x, s_k, wv, ln, hi);
  proj128_lds(Wv, bv, s_x, s_v, wv, ln, hi);
  __syncthreads();

  // ---- Phase 2: per-head attention ----
  const float scale = 0.17677669529663687f;  // 1/sqrt(32)
  for (int h = 0; h < NHEAD; ++h) {
    const int hoff = h * HD;

    // scores[q][k] = sum_d Q[q][hoff+d] * K[k][hoff+d]   (M=96,N=96,K=32 -> 36 tiles)
    for (int t = wv; t < 36; t += 8) {
      const int m0   = (t / 6) * 16;
      const int n0   = (t % 6) * 16;
      const int ncol = n0 + ln;
      v8f c;
#pragma unroll
      for (int r = 0; r < 8; ++r) c[r] = 0.f;
      v16bf a = ld_a_bf(s_q + (m0 + ln) * C_DIM, hoff, hi);
      v16bf b = ld_b_contig_bf(s_k + ncol * C_DIM + hoff + hi * 16);
      c = wmma_bf16(a, b, c);
      const int mrow = m0 + hi * 8;
#pragma unroll
      for (int r = 0; r < 8; ++r) s_sc[(mrow + r) * T_DIM + ncol] = c[r] * scale;
    }
    __syncthreads();

    // softmax over the QUERY axis (per key column); stride-96 f32 -> conflict free
    if (tid < T_DIM) {
      float mx = -3.0e38f;
      for (int q = 0; q < T_DIM; ++q) mx = fmaxf(mx, s_sc[q * T_DIM + tid]);
      float sum = 0.f;
      for (int q = 0; q < T_DIM; ++q) {
        float e = __expf(s_sc[q * T_DIM + tid] - mx);
        s_sc[q * T_DIM + tid] = e;
        sum += e;
      }
      const float rs = 1.0f / sum;
      for (int q = 0; q < T_DIM; ++q) s_sc[q * T_DIM + tid] *= rs;
    }
    __syncthreads();

    // out[q][d] = sum_k attn[q][k] * V[k][hoff+d]   (M=96,N=32,K=96 -> 12 tiles x 3 K-steps)
    for (int t = wv; t < 12; t += 8) {
      const int m0   = (t >> 1) * 16;
      const int n0   = (t & 1) * 16;
      const int dcol = n0 + ln;
      v8f c;
#pragma unroll
      for (int r = 0; r < 8; ++r) c[r] = 0.f;
      const float* arow = s_sc + (m0 + ln) * T_DIM;
#pragma unroll
      for (int kk = 0; kk < T_DIM; kk += 32) {
        v16bf a = ld_a_f32(arow, kk, hi);
        v16bf b = ld_b_strided_bf(s_v + (kk + hi * 16) * C_DIM + hoff + dcol, C_DIM);
        c = wmma_bf16(a, b, c);
      }
      const int mrow = m0 + hi * 8;
#pragma unroll
      for (int r = 0; r < 8; ++r)
        s_x[(mrow + r) * C_DIM + hoff + dcol] = (__bf16)c[r];  // attn-out overwrites x
    }
    __syncthreads();
  }

  // ---- Phase 3: out = attn_out @ Wo^T + bo -> global fp32 ----
  for (int t = wv; t < 48; t += 8) {
    const int m0  = (t >> 3) * 16;
    const int n0  = (t & 7) * 16;
    const int col = n0 + ln;
    const float bval = bo[col];
    v8f c;
#pragma unroll
    for (int r = 0; r < 8; ++r) c[r] = bval;
    const __bf16* arow = s_x + (m0 + ln) * C_DIM;
    const float*  wrow = Wo + (long)col * C_DIM;
#pragma unroll
    for (int kk = 0; kk < C_DIM; kk += 32) {
      v16bf a = ld_a_bf(arow, kk, hi);
      v16bf b = ld_b_contig_f32(wrow + kk + hi * 16);
      c = wmma_bf16(a, b, c);
    }
    const int mrow = m0 + hi * 8;
#pragma unroll
    for (int r = 0; r < 8; ++r) ob[(mrow + r) * C_DIM + col] = c[r];
  }
}

extern "C" void kernel_launch(void* const* d_in, const int* in_sizes, int n_in,
                              void* d_out, int out_size, void* d_ws, size_t ws_size,
                              hipStream_t stream) {
  const float* x  = (const float*)d_in[0];
  const float* Wq = (const float*)d_in[1];
  const float* bq = (const float*)d_in[2];
  const float* Wk = (const float*)d_in[3];
  const float* bk = (const float*)d_in[4];
  const float* Wv = (const float*)d_in[5];
  const float* bv = (const float*)d_in[6];
  const float* Wo = (const float*)d_in[7];
  const float* bo = (const float*)d_in[8];
  // d_in[9] = node_num: attention is independent per node, grouping doesn't
  // change the memory layout of the computation.
  float* out = (float*)d_out;

  const int nodes = in_sizes[0] / (T_DIM * C_DIM);  // B*N = 4096
  spatial_attn_fused<<<nodes, 256, SMEM_BYTES, stream>>>(x, Wq, bq, Wk, bk, Wv, bv,
                                                         Wo, bo, out);
}
